// DeepseekMlpBlock_13889924235945
// MI455X (gfx1250) — compile-verified, matrix operand order checked by reading, not run on hardware
//
#include <hip/hip_runtime.h>

typedef _Float16 half_t;
typedef __attribute__((ext_vector_type(16))) _Float16 v16h;
typedef __attribute__((ext_vector_type(8)))  _Float16 v8h;
typedef __attribute__((ext_vector_type(8)))  float    v8f;
typedef __attribute__((ext_vector_type(4)))  float    vf4;
typedef __attribute__((ext_vector_type(4)))  int      vi4;

#define H_DIM 4096
#define I_DIM 14336
#define KS 64          // K-slice staged in LDS per iteration
#define LDS_STRIDE 72  // KS + 8 halfs padding (144 B row stride, 16B-aligned)

// D = A(16x32 f16) * B(32x16 f16) + C(16x16 f32)
__device__ __forceinline__ v8f wmma16(v16h a, v16h b, v8f c) {
  return __builtin_amdgcn_wmma_f32_16x16x32_f16(
      /*neg_a=*/false, a, /*neg_b=*/false, b,
      /*c_mod=*/(short)0, c, /*reuse_a=*/false, /*reuse_b=*/false);
}

// Build a 16-bit A/B fragment for one 16x16(x32) tile from an LDS row.
// Per ISA 7.12.2: lanes 0-15 / 16-31 split K groups of 8 at K=0/8 and K=16/24.
// Caller passes k0 = ks*32 + (lane>=16 ? 8 : 0); elements 8..15 come from k0+16.
__device__ __forceinline__ v16h frag_load(const half_t* __restrict__ row, int k0) {
  v8h lo = *(const v8h*)(row + k0);
  v8h hi = *(const v8h*)(row + k0 + 16);
  v16h r;
#pragma unroll
  for (int i = 0; i < 8; ++i) { r[i] = lo[i]; r[i + 8] = hi[i]; }
  return r;
}

// ---------------------------------------------------------------------------
// Phase 1: h = silu(x @ dq(w0,s0)) * (x @ dq(w1,s1)),  h stored f16 in ws
// Grid: (I/128, M/128), block 256 (8 wave32s in 2(M)x4(N) arrangement)
// ---------------------------------------------------------------------------
__global__ __launch_bounds__(256) void mlp_gateup_kernel(
    const float* __restrict__ x, const int* __restrict__ w0,
    const int* __restrict__ w1, const float* __restrict__ s0,
    const float* __restrict__ s1, half_t* __restrict__ h) {
  __shared__ alignas(16) half_t Xs[128][LDS_STRIDE];
  __shared__ alignas(16) half_t W0s[128][LDS_STRIDE];  // transposed: [n][k]
  __shared__ alignas(16) half_t W1s[128][LDS_STRIDE];

  const int tid   = threadIdx.x;
  const int lane  = tid & 31;
  const int wave  = tid >> 5;
  const int n0    = blockIdx.x * 128;
  const int m0    = blockIdx.y * 128;
  const int wm    = (wave >> 2) * 64;   // wave M offset: 0 / 64
  const int wn    = (wave & 3) * 32;    // wave N offset: 0/32/64/96
  const int lidx  = lane & 15;
  const int lhalf = lane >> 4;

  // loader mappings
  const int xc = (tid & 15) * 4;  // x: float4 per thread, 16 thr/row
  const int xr = tid >> 4;        // 0..15
  const int wc = (tid & 31) * 4;  // w: int4 per thread, 32 thr/krow
  const int wr = tid >> 5;        // 0..7

  v8f zero = {};
  v8f accg[4][2], accu[4][2];
#pragma unroll
  for (int mi = 0; mi < 4; ++mi)
#pragma unroll
    for (int ni = 0; ni < 2; ++ni) { accg[mi][ni] = zero; accu[mi][ni] = zero; }

  const int nstages = H_DIM / KS;  // 64
  for (int st = 0; st < nstages; ++st) {
    const int kg = st * KS;        // global K base
    const int kb = kg >> 7;        // 128-row scale-block index

    if (st + 1 < nstages) {        // gfx1250 global_prefetch_b8 of next slice
      __builtin_prefetch(&w0[(size_t)(kg + KS) * I_DIM + n0 + wc], 0, 1);
      __builtin_prefetch(&w1[(size_t)(kg + KS) * I_DIM + n0 + wc], 0, 1);
    }

    __syncthreads();
    // stage x tile: 128 rows x KS cols, f32 -> f16
#pragma unroll
    for (int rr = 0; rr < 8; ++rr) {
      const int row = rr * 16 + xr;
      vf4 v = *(const vf4*)&x[(size_t)(m0 + row) * H_DIM + kg + xc];
      Xs[row][xc + 0] = (half_t)v.x;
      Xs[row][xc + 1] = (half_t)v.y;
      Xs[row][xc + 2] = (half_t)v.z;
      Xs[row][xc + 3] = (half_t)v.w;
    }
    // stage w0/w1: dequant (scale folded in) + transpose to [n][k]
    vf4 sc0 = *(const vf4*)&s0[(size_t)kb * I_DIM + n0 + wc];
    vf4 sc1 = *(const vf4*)&s1[(size_t)kb * I_DIM + n0 + wc];
#pragma unroll
    for (int rr = 0; rr < 8; ++rr) {
      const int krow = rr * 8 + wr;
      vi4 a = *(const vi4*)&w0[(size_t)(kg + krow) * I_DIM + n0 + wc];
      vi4 b = *(const vi4*)&w1[(size_t)(kg + krow) * I_DIM + n0 + wc];
      W0s[wc + 0][krow] = (half_t)((float)a.x * sc0.x);
      W0s[wc + 1][krow] = (half_t)((float)a.y * sc0.y);
      W0s[wc + 2][krow] = (half_t)((float)a.z * sc0.z);
      W0s[wc + 3][krow] = (half_t)((float)a.w * sc0.w);
      W1s[wc + 0][krow] = (half_t)((float)b.x * sc1.x);
      W1s[wc + 1][krow] = (half_t)((float)b.y * sc1.y);
      W1s[wc + 2][krow] = (half_t)((float)b.z * sc1.z);
      W1s[wc + 3][krow] = (half_t)((float)b.w * sc1.w);
    }
    __syncthreads();

#pragma unroll
    for (int ks = 0; ks < KS / 32; ++ks) {
      const int k0 = ks * 32 + lhalf * 8;
      v16h b0[2], b1[2];
#pragma unroll
      for (int ni = 0; ni < 2; ++ni) {
        const int col = wn + ni * 16 + lidx;
        b0[ni] = frag_load(&W0s[col][0], k0);
        b1[ni] = frag_load(&W1s[col][0], k0);
      }
#pragma unroll
      for (int mi = 0; mi < 4; ++mi) {
        const int row = wm + mi * 16 + lidx;
        v16h a = frag_load(&Xs[row][0], k0);
#pragma unroll
        for (int ni = 0; ni < 2; ++ni) {
          accg[mi][ni] = wmma16(a, b0[ni], accg[mi][ni]);
          accu[mi][ni] = wmma16(a, b1[ni], accu[mi][ni]);
        }
      }
    }
  }

  // epilogue: h = silu(g) * u  (C/D layout: lane%16 = N, +8 rows for lane>=16)
#pragma unroll
  for (int mi = 0; mi < 4; ++mi) {
#pragma unroll
    for (int ni = 0; ni < 2; ++ni) {
      const int col   = n0 + wn + ni * 16 + lidx;
      const int rbase = m0 + wm + mi * 16 + lhalf * 8;
#pragma unroll
      for (int r = 0; r < 8; ++r) {
        float g  = accg[mi][ni][r];
        float u  = accu[mi][ni][r];
        float hv = (g / (1.0f + __expf(-g))) * u;
        h[(size_t)(rbase + r) * I_DIM + col] = (half_t)hv;
      }
    }
  }
}

// ---------------------------------------------------------------------------
// Phase 2: out = h @ dq(w2, s2), f32 output
// A-tile staged via gfx1250 async global->LDS DMA (ASYNCcnt), B-tile dequanted
// through VGPRs (needs transpose + int->f16*scale transform).
// Grid: (H/128, M/128), block 256
// ---------------------------------------------------------------------------
__global__ __launch_bounds__(256) void mlp_down_kernel(
    const half_t* __restrict__ h, const int* __restrict__ w2,
    const float* __restrict__ s2, float* __restrict__ out) {
  __shared__ alignas(16) half_t Hs[128][LDS_STRIDE];
  __shared__ alignas(16) half_t W2s[128][LDS_STRIDE];  // transposed: [n][k]

  const int tid   = threadIdx.x;
  const int lane  = tid & 31;
  const int wave  = tid >> 5;
  const int n0    = blockIdx.x * 128;  // over H
  const int m0    = blockIdx.y * 128;
  const int wm    = (wave >> 2) * 64;
  const int wn    = (wave & 3) * 32;
  const int lidx  = lane & 15;
  const int lhalf = lane >> 4;

  const int hc = (tid & 7) * 8;   // h: one b128 per thread-chunk, 8 thr/row
  const int hr = tid >> 3;        // 0..31
  const int wc = (tid & 31) * 4;
  const int wr = tid >> 5;

  v8f zero = {};
  v8f acc[4][2];
#pragma unroll
  for (int mi = 0; mi < 4; ++mi)
#pragma unroll
    for (int ni = 0; ni < 2; ++ni) acc[mi][ni] = zero;

  const int nstages = I_DIM / KS;  // 224
  for (int st = 0; st < nstages; ++st) {
    const int kg = st * KS;
    const int kb = kg >> 7;

    if (st + 1 < nstages)
      __builtin_prefetch(&w2[(size_t)(kg + KS) * H_DIM + n0 + wc], 0, 1);

    __syncthreads();
    // stage A (h) tile: pure f16 copy -> use async global->LDS b128 DMA.
    // GV addressing: VDST = wave-relative LDS byte offset, VADDR = 64b address.
#pragma unroll
    for (int rr = 0; rr < 4; ++rr) {
      const int row = rr * 32 + hr;
      unsigned long long gaddr = (unsigned long long)(uintptr_t)
          &h[(size_t)(m0 + row) * I_DIM + kg + hc];
      unsigned ldsoff = (unsigned)(uintptr_t)&Hs[row][hc];
      asm volatile("global_load_async_to_lds_b128 %0, %1, off"
                   :: "v"(ldsoff), "v"(gaddr) : "memory");
    }
    // stage B (w2) tile: dequant + transpose through VGPRs (DScnt path)
    vf4 sc = *(const vf4*)&s2[(size_t)kb * H_DIM + n0 + wc];
#pragma unroll
    for (int rr = 0; rr < 8; ++rr) {
      const int krow = rr * 8 + wr;
      vi4 a = *(const vi4*)&w2[(size_t)(kg + krow) * H_DIM + n0 + wc];
      W2s[wc + 0][krow] = (half_t)((float)a.x * sc.x);
      W2s[wc + 1][krow] = (half_t)((float)a.y * sc.y);
      W2s[wc + 2][krow] = (half_t)((float)a.z * sc.z);
      W2s[wc + 3][krow] = (half_t)((float)a.w * sc.w);
    }
    // drain this wave's async DMAs (barrier below only covers DScnt)
    asm volatile("s_wait_asynccnt 0x0" ::: "memory");
    __syncthreads();

#pragma unroll
    for (int ks = 0; ks < KS / 32; ++ks) {
      const int k0 = ks * 32 + lhalf * 8;
      v16h bf[2];
#pragma unroll
      for (int ni = 0; ni < 2; ++ni) {
        const int col = wn + ni * 16 + lidx;
        bf[ni] = frag_load(&W2s[col][0], k0);
      }
#pragma unroll
      for (int mi = 0; mi < 4; ++mi) {
        const int row = wm + mi * 16 + lidx;
        v16h a = frag_load(&Hs[row][0], k0);
#pragma unroll
        for (int ni = 0; ni < 2; ++ni)
          acc[mi][ni] = wmma16(a, bf[ni], acc[mi][ni]);
      }
    }
  }

#pragma unroll
  for (int mi = 0; mi < 4; ++mi) {
#pragma unroll
    for (int ni = 0; ni < 2; ++ni) {
      const int col   = n0 + wn + ni * 16 + lidx;
      const int rbase = m0 + wm + mi * 16 + lhalf * 8;
#pragma unroll
      for (int r = 0; r < 8; ++r)
        out[(size_t)(rbase + r) * H_DIM + col] = acc[mi][ni][r];
    }
  }
}

extern "C" void kernel_launch(void* const* d_in, const int* in_sizes, int n_in,
                              void* d_out, int out_size, void* d_ws, size_t ws_size,
                              hipStream_t stream) {
  (void)in_sizes; (void)n_in; (void)out_size; (void)ws_size;
  const float* x  = (const float*)d_in[0];
  const int*   w0 = (const int*)d_in[1];
  const int*   w1 = (const int*)d_in[2];
  const int*   w2 = (const int*)d_in[3];
  const float* s0 = (const float*)d_in[4];
  const float* s1 = (const float*)d_in[5];
  const float* s2 = (const float*)d_in[6];
  // d_in[7] = blocksize (128), compile-time constant here

  half_t* hbuf = (half_t*)d_ws;          // M x I f16 intermediate (~235 MB)
  float*  outp = (float*)d_out;          // M x H f32

  dim3 block(256);
  dim3 grid1(I_DIM / 128, (4 * 2048) / 128);  // 112 x 64
  dim3 grid2(H_DIM / 128, (4 * 2048) / 128);  // 32 x 64
  mlp_gateup_kernel<<<grid1, block, 0, stream>>>(x, w0, w1, s0, s1, hbuf);
  mlp_down_kernel<<<grid2, block, 0, stream>>>(hbuf, w2, s2, outp);
}
